// MultiHeadAttention_3959959847654
// MI455X (gfx1250) — matrix operand order repaired and checked by reading, not compiled, not run on hardware
//
#include <hip/hip_runtime.h>

typedef __attribute__((ext_vector_type(16))) __bf16 v16bf;
typedef __attribute__((ext_vector_type(8)))  __bf16 v8bf;
typedef __attribute__((ext_vector_type(8)))  float  v8f;
typedef __attribute__((ext_vector_type(4)))  unsigned short u16x4;
typedef __attribute__((ext_vector_type(4)))  unsigned int   u32x4;
typedef __attribute__((ext_vector_type(8)))  unsigned int   u32x8;

#define Bsz  16
#define Nseq 1024
#define Edim 512
#define Hn   8
#define Dh   512
#define HIDD 4096
#define TOK  (Bsz*Nseq)

__device__ __forceinline__ unsigned short f2bf(float f) {
  unsigned u = __float_as_uint(f);
  u += 0x7FFFu + ((u >> 16) & 1u);          // round-to-nearest-even
  return (unsigned short)(u >> 16);
}

union FragU { v16bf v; v8bf h[2]; };

// TDM: 2-D block copy, 128 rows x 32 bf16, row stride lda elems, into LDS.
// D# group0: count=1, lds_addr, global_addr, type=2("image").
// D# group1: data_size=2B, tensor dims == tile dims (no OOB), stride0=lda.
__device__ __forceinline__ void tdm_load_A_tile(const unsigned short* gsrc,
                                                long lda, unsigned lds_byte) {
  const unsigned long gaddr = (unsigned long)gsrc;
  u32x4 g0;
  g0[0] = 1u;
  g0[1] = lds_byte;
  g0[2] = (unsigned)gaddr;
  g0[3] = (unsigned)(gaddr >> 32) | 0x80000000u;
  u32x8 g1;
  g1[0] = (1u << 16);    // data_size = 2 bytes
  g1[1] = (32u << 16);   // tensor_dim0 = 32
  g1[2] = (128u << 16);  // tensor_dim1 = 128
  g1[3] = (32u << 16);   // tile_dim0 = 32
  g1[4] = 128u;          // tile_dim1 = 128, tile_dim2 = 0
  g1[5] = (unsigned)lda; // tensor_dim0_stride[31:0]
  g1[6] = 0u;
  g1[7] = 0u;
  asm volatile("tensor_load_to_lds %0, %1" :: "s"(g0), "s"(g1) : "memory");
}

// ---------------------------------------------------------------- convert
__global__ void cvt_bf16_kernel(const float* __restrict__ in,
                                unsigned short* __restrict__ out, int n4) {
  int i = blockIdx.x * blockDim.x + threadIdx.x;
  if (i >= n4) return;
  float4 v = ((const float4*)in)[i];
  u16x4 o;
  o[0] = f2bf(v.x); o[1] = f2bf(v.y); o[2] = f2bf(v.z); o[3] = f2bf(v.w);
  ((u16x4*)out)[i] = o;
}

// ---------------------------------------------------------------- generic NN GEMM
// C[M,N] = A[M,K](bf16) * B[K,N](bf16); 128x128 tile, BK=32, 256 thr = 8 waves.
// Double-buffered LDS: TDM prefetches A tile k+1 while WMMAs chew tile k;
// the VALU/DS path transposes B tile k+1 behind the WMMAs; joined by
// s_wait_tensorcnt + barrier at the end of each iteration.
template <bool F32OUT>
__global__ __launch_bounds__(256) void gemm_bf16_nn(
    const unsigned short* __restrict__ A, long lda,
    const unsigned short* __restrict__ Bm, long ldb,
    void* __restrict__ Cp, long ldc,
    int K, int Hdiv,
    long sAb, long sAh, long sBb, long sBh, long sCb, long sCh,
    const float* __restrict__ bias) {
  extern __shared__ unsigned short smem[];   // 2 x (As 4096 + Bs 4096) elems
  // buffer p: As at elem p*8192 (LDS byte p*16384), Bs at elem p*8192+4096

  const int z  = blockIdx.z;
  const int zb = z / Hdiv, zh = z % Hdiv;
  A  += (long)zb * sAb + (long)zh * sAh;
  Bm += (long)zb * sBb + (long)zh * sBh;
  const long coff = (long)zb * sCb + (long)zh * sCh;

  const int m0 = blockIdx.y * 128, n0 = blockIdx.x * 128;
  const int tid = threadIdx.x, lane = tid & 31, wave = tid >> 5;
  const int wm = wave >> 2, wn = wave & 3;
  const int hf = lane >> 4, ln = lane & 15;
  const int kr = tid >> 3, nc = (tid & 7) * 16;   // B-tile loader coords

  v8f acc[4][2] = {};

  // ---- prologue: stage tile 0 into buffer 0
  if (wave == 0) tdm_load_A_tile(A + (long)m0 * lda, lda, 0);
  {
    const unsigned short* s = Bm + (long)kr * ldb + n0 + nc;
    union { uint4 q[2]; unsigned short u[16]; } tb;
    tb.q[0] = *(const uint4*)s;
    tb.q[1] = *(const uint4*)(s + 8);
    unsigned short* Bs = smem + 4096;
#pragma unroll
    for (int i = 0; i < 16; ++i) Bs[(nc + i) * 32 + kr] = tb.u[i];
  }
  if (wave == 0) __builtin_amdgcn_s_wait_tensorcnt(0);
  __syncthreads();

  int cur = 0;
  for (int k0 = 0; k0 < K; k0 += 32) {
    const int nxt = cur ^ 1;
    const bool more = (k0 + 32) < K;

    // issue TDM for next A tile into the other buffer (async, overlaps WMMA)
    if (more && wave == 0)
      tdm_load_A_tile(A + (long)m0 * lda + k0 + 32, lda, (unsigned)(nxt * 16384));

    // ---- compute on current buffer
    {
      const unsigned short* As = smem + cur * 8192;
      const unsigned short* Bs = smem + cur * 8192 + 4096;
      FragU af[4], bf[2];
#pragma unroll
      for (int t = 0; t < 4; ++t) {
        const int row = wm * 64 + t * 16 + ln;
        af[t].h[0] = *(const v8bf*)&As[row * 32 + hf * 8];
        af[t].h[1] = *(const v8bf*)&As[row * 32 + hf * 8 + 16];
      }
#pragma unroll
      for (int t = 0; t < 2; ++t) {
        const int col = wn * 32 + t * 16 + ln;
        bf[t].h[0] = *(const v8bf*)&Bs[col * 32 + hf * 8];
        bf[t].h[1] = *(const v8bf*)&Bs[col * 32 + hf * 8 + 16];
      }
#pragma unroll
      for (int i = 0; i < 4; ++i)
#pragma unroll
        for (int j = 0; j < 2; ++j)
          acc[i][j] = __builtin_amdgcn_wmma_f32_16x16x32_bf16(
              false, af[i].v, false, bf[j].v, (short)0, acc[i][j], false, false);
    }

    // ---- stage next B tile (transposed) into the other buffer, behind WMMAs
    if (more) {
      const unsigned short* s = Bm + (long)(k0 + 32 + kr) * ldb + n0 + nc;
      union { uint4 q[2]; unsigned short u[16]; } tb;
      tb.q[0] = *(const uint4*)s;
      tb.q[1] = *(const uint4*)(s + 8);
      unsigned short* Bs = smem + nxt * 8192 + 4096;
#pragma unroll
      for (int i = 0; i < 16; ++i) Bs[(nc + i) * 32 + kr] = tb.u[i];
      if (k0 + 64 < K)  // warm tile k+2 (global_prefetch_b8)
        __builtin_prefetch(Bm + (long)(k0 + 64 + kr) * ldb + n0 + nc, 0, 1);
    }

    if (more && wave == 0) __builtin_amdgcn_s_wait_tensorcnt(0);
    __syncthreads();
    cur = nxt;
  }

#pragma unroll
  for (int i = 0; i < 4; ++i) {
#pragma unroll
    for (int j = 0; j < 2; ++j) {
      const int col = n0 + wn * 32 + j * 16 + ln;
      float bv = 0.f;
      if constexpr (F32OUT) bv = bias[col];
#pragma unroll
      for (int r = 0; r < 8; ++r) {
        const long m = m0 + wm * 64 + i * 16 + r + 8 * hf;
        if constexpr (F32OUT)
          ((float*)Cp)[coff + m * ldc + col] = acc[i][j][r] + bv;
        else
          ((unsigned short*)Cp)[coff + m * ldc + col] = f2bf(acc[i][j][r]);
      }
    }
  }
}

// ---------------------------------------------------------------- fused QK^T + mask + softmax -> P(bf16)
// One workgroup = 32 query rows x 1024 kv cols of one (b,h). 8 waves x 128 cols.
// 2 row-groups per wave: every K fragment feeds 2 WMMAs, so the scheduler
// keeps all 8 fragments live and overlaps the load clause with matrix ops.
__global__ __launch_bounds__(256) void attn_score_softmax(
    const unsigned short* __restrict__ Qh, const unsigned short* __restrict__ Kh,
    const float* __restrict__ mask, unsigned short* __restrict__ P) {
  __shared__ float red[8][32];
  __shared__ float rowred[32];

  const int bh = blockIdx.y;
  const int b = bh >> 3, h = bh & 7;
  const int r0 = blockIdx.x * 32;
  const unsigned short* Qb = Qh + ((long)(b * Nseq + r0)) * HIDD + h * Dh;
  const unsigned short* Kb = Kh + ((long)b * Nseq) * HIDD + h * Dh;

  const int tid = threadIdx.x, lane = tid & 31, wave = tid >> 5;
  const int hf = lane >> 4, ln = lane & 15;
  const int cbase = wave * 128;

  v8f acc[2][8] = {};
  for (int k0 = 0; k0 < Dh; k0 += 32) {
    FragU af[2], bf[8];
#pragma unroll
    for (int g = 0; g < 2; ++g) {
      const long row = g * 16 + ln;
      af[g].h[0] = *(const v8bf*)&Qb[row * HIDD + k0 + hf * 8];
      af[g].h[1] = *(const v8bf*)&Qb[row * HIDD + k0 + hf * 8 + 16];
    }
#pragma unroll
    for (int t = 0; t < 8; ++t) {
      const long col = cbase + t * 16 + ln;
      bf[t].h[0] = *(const v8bf*)&Kb[col * HIDD + k0 + hf * 8];
      bf[t].h[1] = *(const v8bf*)&Kb[col * HIDD + k0 + hf * 8 + 16];
    }
#pragma unroll
    for (int g = 0; g < 2; ++g)
#pragma unroll
      for (int t = 0; t < 8; ++t)
        acc[g][t] = __builtin_amdgcn_wmma_f32_16x16x32_bf16(
            false, af[g].v, false, bf[t].v, (short)0, acc[g][t], false, false);
  }

  const float scale = 0.04419417382415922f;  // 1/sqrt(512)
  float mv[8];
#pragma unroll
  for (int t = 0; t < 8; ++t) mv[t] = mask[b * Nseq + cbase + t * 16 + ln];
#pragma unroll
  for (int g = 0; g < 2; ++g)
#pragma unroll
    for (int t = 0; t < 8; ++t)
#pragma unroll
      for (int r = 0; r < 8; ++r) acc[g][t][r] = acc[g][t][r] * scale + mv[t];

  // per-lane row maxima; rows m = g*16 + r + 8*hf
  float rmax[2][8];
#pragma unroll
  for (int g = 0; g < 2; ++g)
#pragma unroll
    for (int r = 0; r < 8; ++r) {
      float m = acc[g][0][r];
#pragma unroll
      for (int t = 1; t < 8; ++t) m = fmaxf(m, acc[g][t][r]);
      rmax[g][r] = m;
    }
#pragma unroll
  for (int off = 1; off < 16; off <<= 1)
#pragma unroll
    for (int g = 0; g < 2; ++g)
#pragma unroll
      for (int r = 0; r < 8; ++r)
        rmax[g][r] = fmaxf(rmax[g][r], __shfl_xor(rmax[g][r], off, 32));

  if (ln == 0)
#pragma unroll
    for (int g = 0; g < 2; ++g)
#pragma unroll
      for (int r = 0; r < 8; ++r) red[wave][g * 16 + r + 8 * hf] = rmax[g][r];
  __syncthreads();
  if (tid < 32) {
    float m = red[0][tid];
#pragma unroll
    for (int w = 1; w < 8; ++w) m = fmaxf(m, red[w][tid]);
    rowred[tid] = m;
  }
  __syncthreads();
  float rm[2][8];
#pragma unroll
  for (int g = 0; g < 2; ++g)
#pragma unroll
    for (int r = 0; r < 8; ++r) rm[g][r] = rowred[g * 16 + r + 8 * hf];

  // exp + row sums
  float rsum[2][8];
#pragma unroll
  for (int g = 0; g < 2; ++g)
#pragma unroll
    for (int r = 0; r < 8; ++r) rsum[g][r] = 0.f;
#pragma unroll
  for (int g = 0; g < 2; ++g)
#pragma unroll
    for (int t = 0; t < 8; ++t)
#pragma unroll
      for (int r = 0; r < 8; ++r) {
        float e = __expf(acc[g][t][r] - rm[g][r]);
        acc[g][t][r] = e;
        rsum[g][r] += e;
      }
#pragma unroll
  for (int off = 1; off < 16; off <<= 1)
#pragma unroll
    for (int g = 0; g < 2; ++g)
#pragma unroll
      for (int r = 0; r < 8; ++r) rsum[g][r] += __shfl_xor(rsum[g][r], off, 32);

  if (ln == 0)
#pragma unroll
    for (int g = 0; g < 2; ++g)
#pragma unroll
      for (int r = 0; r < 8; ++r) red[wave][g * 16 + r + 8 * hf] = rsum[g][r];
  __syncthreads();
  if (tid < 32) {
    float s = red[0][tid];
#pragma unroll
    for (int w = 1; w < 8; ++w) s += red[w][tid];
    rowred[tid] = s;
  }
  __syncthreads();
  float rinv[2][8];
#pragma unroll
  for (int g = 0; g < 2; ++g)
#pragma unroll
    for (int r = 0; r < 8; ++r) rinv[g][r] = 1.f / rowred[g * 16 + r + 8 * hf];

  unsigned short* Pb = P + ((long)bh * Nseq + r0) * Nseq;
#pragma unroll
  for (int g = 0; g < 2; ++g)
#pragma unroll
    for (int t = 0; t < 8; ++t) {
      const int col = cbase + t * 16 + ln;
#pragma unroll
      for (int r = 0; r < 8; ++r) {
        const int m = g * 16 + r + 8 * hf;
        Pb[(long)m * Nseq + col] = f2bf(acc[g][t][r] * rinv[g][r]);
      }
    }
}

// ---------------------------------------------------------------- launcher
extern "C" void kernel_launch(void* const* d_in, const int* in_sizes, int n_in,
                              void* d_out, int out_size, void* d_ws, size_t ws_size,
                              hipStream_t stream) {
  const float* q    = (const float*)d_in[0];
  const float* k    = (const float*)d_in[1];
  const float* v    = (const float*)d_in[2];
  const float* mask = (const float*)d_in[3];
  const float* Wq   = (const float*)d_in[4];
  const float* Wk   = (const float*)d_in[5];
  const float* Wv   = (const float*)d_in[6];
  const float* Wo   = (const float*)d_in[7];
  const float* bo   = (const float*)d_in[8];
  float* out = (float*)d_out;

  char* w = (char*)d_ws;
  const size_t SZ_TOK = (size_t)TOK * Edim * 2;             // 16 MB
  const size_t SZ_W   = (size_t)Edim * HIDD * 2;            // 4 MB
  const size_t SZ_BIG = (size_t)TOK * HIDD * 2;             // 128 MB
  const size_t SZ_P   = (size_t)Bsz * Hn * Nseq * Nseq * 2; // 256 MB
  unsigned short* qb  = (unsigned short*)w; w += SZ_TOK;
  unsigned short* kb  = (unsigned short*)w; w += SZ_TOK;
  unsigned short* vb  = (unsigned short*)w; w += SZ_TOK;
  unsigned short* wqb = (unsigned short*)w; w += SZ_W;
  unsigned short* wkb = (unsigned short*)w; w += SZ_W;
  unsigned short* wvb = (unsigned short*)w; w += SZ_W;
  unsigned short* wob = (unsigned short*)w; w += SZ_W;
  unsigned short* Qh  = (unsigned short*)w; w += SZ_BIG;
  unsigned short* Kh  = (unsigned short*)w; w += SZ_BIG;
  unsigned short* Vh  = (unsigned short*)w; w += SZ_BIG;
  unsigned short* P   = (unsigned short*)w; w += SZ_P;
  unsigned short* AO  = (unsigned short*)w; w += SZ_BIG;

  const int LDSB = 2 * 2 * 128 * 32 * 2;   // double-buffered As+Bs, bytes

  // 1) fp32 -> bf16
  {
    int n4 = TOK * Edim / 4;
    cvt_bf16_kernel<<<n4 / 256, 256, 0, stream>>>(q, qb, n4);
    cvt_bf16_kernel<<<n4 / 256, 256, 0, stream>>>(k, kb, n4);
    cvt_bf16_kernel<<<n4 / 256, 256, 0, stream>>>(v, vb, n4);
    int w4 = Edim * HIDD / 4;
    cvt_bf16_kernel<<<w4 / 256, 256, 0, stream>>>(Wq, wqb, w4);
    cvt_bf16_kernel<<<w4 / 256, 256, 0, stream>>>(Wk, wkb, w4);
    cvt_bf16_kernel<<<w4 / 256, 256, 0, stream>>>(Wv, wvb, w4);
    cvt_bf16_kernel<<<w4 / 256, 256, 0, stream>>>(Wo, wob, w4);
  }

  // 2) QKV projections: [TOK,512] x [512,4096] -> [TOK,4096] bf16
  {
    dim3 g(HIDD / 128, TOK / 128, 1);
    gemm_bf16_nn<false><<<g, 256, LDSB, stream>>>(qb, Edim, wqb, HIDD, Qh, HIDD,
                                                  Edim, 1, 0, 0, 0, 0, 0, 0, nullptr);
    gemm_bf16_nn<false><<<g, 256, LDSB, stream>>>(kb, Edim, wkb, HIDD, Kh, HIDD,
                                                  Edim, 1, 0, 0, 0, 0, 0, 0, nullptr);
    gemm_bf16_nn<false><<<g, 256, LDSB, stream>>>(vb, Edim, wvb, HIDD, Vh, HIDD,
                                                  Edim, 1, 0, 0, 0, 0, 0, 0, nullptr);
  }

  // 3) fused scores + mask + softmax -> P (bf16)
  attn_score_softmax<<<dim3(Nseq / 32, Bsz * Hn), 256, 0, stream>>>(Qh, Kh, mask, P);

  // 4) P x V -> AO (bf16), batched over 128 (b,h)
  {
    dim3 g(Dh / 128, Nseq / 128, Bsz * Hn);
    gemm_bf16_nn<false><<<g, 256, LDSB, stream>>>(
        P, Nseq, Vh, HIDD, AO, HIDD, Nseq, Hn,
        (long)Hn * Nseq * Nseq, (long)Nseq * Nseq,
        (long)Nseq * HIDD, (long)Dh,
        (long)Nseq * HIDD, (long)Dh, nullptr);
  }

  // 5) AO x Wo + bo -> out (fp32)
  {
    dim3 g(Edim / 128, TOK / 128, 1);
    gemm_bf16_nn<true><<<g, 256, LDSB, stream>>>(AO, HIDD, wob, Edim, out, Edim,
                                                 HIDD, 1, 0, 0, 0, 0, 0, 0, bo);
  }
}
